// TransformerEncoder_51419348468311
// MI455X (gfx1250) — matrix operand order repaired and checked by reading, not compiled, not run on hardware
//
#include <hip/hip_runtime.h>
#include <hip/hip_bf16.h>
#include <math.h>

// ---------------------------------------------------------------------------
// Types for CDNA5 WMMA
// ---------------------------------------------------------------------------
typedef __attribute__((ext_vector_type(16))) __bf16 v16bf;
typedef __attribute__((ext_vector_type(8)))  float  v8f;
typedef int i32x4 __attribute__((ext_vector_type(4)));

#define F_DIM  1024
#define HEADS  16
#define DH     64
#define MLPH   4096
#define BATCH  2
#define TSEQ   2048
#define NTOK   (BATCH * TSEQ)   // 4096 rows total

// ---------------------------------------------------------------------------
// CDNA5 async global->LDS copy (ASYNCcnt path), with portable fallback.
// ---------------------------------------------------------------------------
#if defined(__has_builtin)
#  if __has_builtin(__builtin_amdgcn_global_load_async_to_lds_b128)
#    define HAVE_ASYNC_LDS 1
#  endif
#endif

#ifdef HAVE_ASYNC_LDS
#  define ASYNC_CP16(gptr, lptr)                                              \
      __builtin_amdgcn_global_load_async_to_lds_b128(                         \
          (__attribute__((address_space(1))) i32x4*)(const void*)(gptr),      \
          (__attribute__((address_space(3))) i32x4*)(void*)(lptr), 0, 0)
#  if __has_builtin(__builtin_amdgcn_s_wait_asynccnt)
#    define WAIT_ASYNC() __builtin_amdgcn_s_wait_asynccnt(0)
#  else
#    define WAIT_ASYNC() asm volatile("s_wait_asynccnt 0x0" ::: "memory")
#  endif
#else
#  define ASYNC_CP16(gptr, lptr)                                              \
      do { *(i32x4*)(void*)(lptr) = *(const i32x4*)(const void*)(gptr); } while (0)
#  define WAIT_ASYNC() do {} while (0)
#endif

__device__ __forceinline__ float gelu_exact(float x) {
    return 0.5f * x * (1.0f + erff(x * 0.7071067811865476f));
}

// ---------------------------------------------------------------------------
// Cast fp32 weight [K, N] -> bf16 transposed [N, K]  (B-operand friendly)
// ---------------------------------------------------------------------------
__global__ void k_cast_transpose(const float* __restrict__ src,
                                 __bf16* __restrict__ dst, int K, int N) {
    size_t idx = (size_t)blockIdx.x * blockDim.x + threadIdx.x;
    size_t total = (size_t)K * N;
    if (idx < total) {
        int k = (int)(idx / N);
        int n = (int)(idx % N);
        dst[(size_t)n * K + k] = (__bf16)src[idx];
    }
}

// ---------------------------------------------------------------------------
// LayerNorm (fp32 in) -> bf16 out.  One block per row.
// ---------------------------------------------------------------------------
__global__ __launch_bounds__(256) void k_layernorm(const float* __restrict__ x,
                                                   const float* __restrict__ g,
                                                   const float* __restrict__ b,
                                                   __bf16* __restrict__ out,
                                                   int F) {
    int row = blockIdx.x;
    const float* xr = x + (size_t)row * F;
    float s = 0.f, s2 = 0.f;
    for (int i = threadIdx.x; i < F; i += blockDim.x) {
        float v = xr[i];
        s += v; s2 += v * v;
    }
    __shared__ float sh1[256], sh2[256];
    sh1[threadIdx.x] = s; sh2[threadIdx.x] = s2;
    __syncthreads();
    for (int o = 128; o > 0; o >>= 1) {
        if (threadIdx.x < (unsigned)o) {
            sh1[threadIdx.x] += sh1[threadIdx.x + o];
            sh2[threadIdx.x] += sh2[threadIdx.x + o];
        }
        __syncthreads();
    }
    float mu  = sh1[0] / F;
    float var = sh2[0] / F - mu * mu;
    float rs  = rsqrtf(var + 1e-5f);
    __bf16* orow = out + (size_t)row * F;
    for (int i = threadIdx.x; i < F; i += blockDim.x) {
        orow[i] = (__bf16)((xr[i] - mu) * rs * g[i] + b[i]);
    }
}

// ---------------------------------------------------------------------------
// WMMA GEMM with double-buffered async LDS staging.
//   C[M,N] = A[M,K](bf16) @ Bt[N,K](bf16)^T + bias, optional GELU / residual.
// Block: 256 threads = 8 waves (2 M x 4 N).  Block tile 128x256, K-stage 64.
// Wave tile 64x64 (4x4 wmma tiles) -> 32 wmma per K-stage per wave.
// LDS: As [2][128][64] + Bs [2][256][64] bf16 = 96 KB.
// 12 async B128 copies per thread per stage.
// ---------------------------------------------------------------------------
template <bool DO_GELU, bool DO_RESID, bool OUT_BF16>
__global__ __launch_bounds__(256) void k_gemm(const __bf16* __restrict__ A,
                                              const __bf16* __restrict__ Bt,
                                              const float*  __restrict__ bias,
                                              const float*  __restrict__ resid,
                                              void* __restrict__ Cout,
                                              int M, int N, int K) {
    const int lane = threadIdx.x & 31;
    const int w    = threadIdx.x >> 5;
    const int wm   = w & 1;        // 0..1
    const int wn   = w >> 1;       // 0..3
    const int lo   = lane & 15;
    const bool hi  = lane >= 16;

    const int mBlock = blockIdx.y * 128;
    const int nBlock = blockIdx.x * 256;

    __shared__ __bf16 As[2][128][64];
    __shared__ __bf16 Bs[2][256][64];

    v8f acc[4][4] = {};

    const int ka = hi ? 8 : 0;   // A-fragment half offset
    const int kb = hi ? 8 : 0;   // B-fragment half offset

    auto issue = [&](int kt, int buf) {
        #pragma unroll
        for (int t = 0; t < 4; ++t) {          // A: 1024 x 16B chunks
            int c   = threadIdx.x + t * 256;
            int row = c >> 3;
            int col = (c & 7) * 8;
            ASYNC_CP16(A + (size_t)(mBlock + row) * K + kt * 64 + col,
                       &As[buf][row][col]);
        }
        #pragma unroll
        for (int t = 0; t < 8; ++t) {          // B: 2048 x 16B chunks
            int c   = threadIdx.x + t * 256;
            int row = c >> 3;
            int col = (c & 7) * 8;
            ASYNC_CP16(Bt + (size_t)(nBlock + row) * K + kt * 64 + col,
                       &Bs[buf][row][col]);
        }
    };

    const int nKT = K / 64;
    issue(0, 0);

    for (int kt = 0; kt < nKT; ++kt) {
        const int buf = kt & 1;
        WAIT_ASYNC();            // our stage-kt copies have landed in LDS
        __syncthreads();         // everyone's copies landed; prev reads done
        if (kt + 1 < nKT) issue(kt + 1, buf ^ 1);

        #pragma unroll
        for (int ks = 0; ks < 2; ++ks) {
            const int k0 = ks * 32;
            // Hoist ALL fragments before the wmma chain so LDS loads pipeline.
            v16bf af[4];
            #pragma unroll
            for (int i = 0; i < 4; ++i) {
                const __bf16* p = &As[buf][wm * 64 + i * 16 + lo][k0 + ka];
                #pragma unroll
                for (int j = 0; j < 8; ++j) { af[i][j] = p[j]; af[i][8 + j] = p[16 + j]; }
            }
            v16bf bfr[4];
            #pragma unroll
            for (int j = 0; j < 4; ++j) {
                const __bf16* p = &Bs[buf][wn * 64 + j * 16 + lo][k0 + kb];
                #pragma unroll
                for (int e = 0; e < 8; ++e) { bfr[j][e] = p[e]; bfr[j][8 + e] = p[16 + e]; }
            }
            #pragma unroll
            for (int i = 0; i < 4; ++i)
                #pragma unroll
                for (int j = 0; j < 4; ++j)
                    acc[i][j] = __builtin_amdgcn_wmma_f32_16x16x32_bf16(
                        false, af[i], false, bfr[j], (short)0, acc[i][j], false, false);
        }
    }

    // Epilogue.  C layout: VGPR r -> row = r + (hi?8:0), col = lo within tile.
    #pragma unroll
    for (int i = 0; i < 4; ++i) {
        #pragma unroll
        for (int j = 0; j < 4; ++j) {
            int col = nBlock + wn * 64 + j * 16 + lo;
            float bv = bias[col];
            #pragma unroll
            for (int r = 0; r < 8; ++r) {
                int row = mBlock + wm * 64 + i * 16 + r + (hi ? 8 : 0);
                float v = acc[i][j][r] + bv;
                if (DO_GELU)  v = gelu_exact(v);
                if (DO_RESID) v += resid[(size_t)row * N + col];
                if (OUT_BF16) ((__bf16*)Cout)[(size_t)row * N + col] = (__bf16)v;
                else          ((float*)Cout)[(size_t)row * N + col] = v;
            }
        }
    }
}

// ---------------------------------------------------------------------------
// Flash-style attention with async LDS staging of the shared K/V block.
// Q,K,V bf16 stored [B,T,F], head h occupies cols h*DH..h*DH+63.
// Scores scaled by 1/sqrt(F)=1/32 (reference scales by sqrt(features)).
// Block = 128 threads = 4 waves; each wave owns 16 query rows; all waves
// stream the same 32-key blocks (K/V tiles staged once per block in LDS).
// Grid: (T/64, B*H).
// ---------------------------------------------------------------------------
__global__ __launch_bounds__(128) void k_attn(const __bf16* __restrict__ Q,
                                              const __bf16* __restrict__ Km,
                                              const __bf16* __restrict__ V,
                                              __bf16* __restrict__ O) {
    const int lane = threadIdx.x & 31;
    const int w    = threadIdx.x >> 5;
    const int lo   = lane & 15;
    const bool hi  = lane >= 16;
    const int bh = blockIdx.y;
    const int b  = bh / HEADS;
    const int h  = bh % HEADS;
    const int q0 = blockIdx.x * 64 + w * 16;

    const size_t headOff = (size_t)h * DH;
    const __bf16* Qbase = Q  + ((size_t)(b * TSEQ + q0)) * F_DIM + headOff;
    const __bf16* Kbase = Km + ((size_t)b * TSEQ) * F_DIM + headOff;
    const __bf16* Vbase = V  + ((size_t)b * TSEQ) * F_DIM + headOff;

    const int ka = hi ? 8 : 0;
    const int kb = hi ? 8 : 0;

    __shared__ __bf16 Kt[2][32][64];
    __shared__ __bf16 Vt[2][32][64];
    __shared__ __bf16 pbuf[4][16][32];

    // Q A-fragments for d=0..31 and d=32..63 (held across the key loop)
    v16bf aq0, aq1;
    {
        const __bf16* p = Qbase + (size_t)lo * F_DIM;
        #pragma unroll
        for (int j = 0; j < 8; ++j) {
            aq0[j]     = p[ka + j];       aq0[8 + j] = p[ka + 16 + j];
            aq1[j]     = p[32 + ka + j];  aq1[8 + j] = p[32 + ka + 16 + j];
        }
    }

    float m[8], l[8];
    v8f o[4] = {};
    #pragma unroll
    for (int r = 0; r < 8; ++r) { m[r] = -1e30f; l[r] = 0.f; }

    const float inv = 0.03125f;  // 1/sqrt(F_DIM)

    auto issue = [&](int blk, int buf) {
        const int key0 = blk * 32;
        #pragma unroll
        for (int t = 0; t < 2; ++t) {
            int c   = threadIdx.x + t * 128;   // 256 chunks of 16B per tile
            int row = c >> 3;
            int col = (c & 7) * 8;
            ASYNC_CP16(Kbase + (size_t)(key0 + row) * F_DIM + col, &Kt[buf][row][col]);
        }
        #pragma unroll
        for (int t = 0; t < 2; ++t) {
            int c   = threadIdx.x + t * 128;
            int row = c >> 3;
            int col = (c & 7) * 8;
            ASYNC_CP16(Vbase + (size_t)(key0 + row) * F_DIM + col, &Vt[buf][row][col]);
        }
    };

    const int nBlk = TSEQ / 32;
    issue(0, 0);

    for (int blk = 0; blk < nBlk; ++blk) {
        const int buf = blk & 1;
        WAIT_ASYNC();
        __syncthreads();
        if (blk + 1 < nBlk) issue(blk + 1, buf ^ 1);

        // ---- scores S[16 x 32] = Q(16x64) @ K^T(64x32) : 4 wmma ----
        // Load ALL K fragments first, then run the wmma chain back-to-back.
        v16bf kb0[2], kb1[2];
        #pragma unroll
        for (int t = 0; t < 2; ++t) {
            const __bf16* kp = &Kt[buf][t * 16 + lo][kb];
            #pragma unroll
            for (int e = 0; e < 8; ++e) { kb0[t][e] = kp[e]; kb0[t][8 + e] = kp[16 + e]; kb1[t][e] = kp[32 + e]; kb1[t][8 + e] = kp[48 + e]; }
        }
        v8f s0 = {}, s1 = {};
        s0 = __builtin_amdgcn_wmma_f32_16x16x32_bf16(false, aq0, false, kb0[0], (short)0, s0, false, false);
        s1 = __builtin_amdgcn_wmma_f32_16x16x32_bf16(false, aq0, false, kb0[1], (short)0, s1, false, false);
        s0 = __builtin_amdgcn_wmma_f32_16x16x32_bf16(false, aq1, false, kb1[0], (short)0, s0, false, false);
        s1 = __builtin_amdgcn_wmma_f32_16x16x32_bf16(false, aq1, false, kb1[1], (short)0, s1, false, false);

        // ---- online softmax (rows live within one 16-lane half) ----
        float p0[8], p1[8];
        #pragma unroll
        for (int r = 0; r < 8; ++r) {
            float v0 = s0[r] * inv, v1 = s1[r] * inv;
            float rm = fmaxf(v0, v1);
            #pragma unroll
            for (int off = 1; off < 16; off <<= 1)
                rm = fmaxf(rm, __shfl_xor(rm, off, 32));
            float mn = fmaxf(m[r], rm);
            float alpha = __expf(m[r] - mn);
            m[r] = mn;
            p0[r] = __expf(v0 - mn);
            p1[r] = __expf(v1 - mn);
            float rs = p0[r] + p1[r];
            #pragma unroll
            for (int off = 1; off < 16; off <<= 1)
                rs += __shfl_xor(rs, off, 32);
            l[r] = l[r] * alpha + rs;
            #pragma unroll
            for (int d = 0; d < 4; ++d) o[d][r] *= alpha;
        }

        // ---- re-fragment P (C-layout -> A-layout) via wave-private LDS.
        // Wave-synchronous: the DS stores (one instruction, all lanes) precede
        // the may-aliasing DS loads; LDS is in-order per wave.
        #pragma unroll
        for (int r = 0; r < 8; ++r) {
            int row = r + (hi ? 8 : 0);
            pbuf[w][row][lo]      = (__bf16)p0[r];
            pbuf[w][row][16 + lo] = (__bf16)p1[r];
        }
        v16bf pa;
        #pragma unroll
        for (int j = 0; j < 8; ++j) {
            pa[j]     = pbuf[w][lo][ka + j];
            pa[8 + j] = pbuf[w][lo][ka + 16 + j];
        }

        // ---- O += P(16x32) @ V(32x64) : 4 wmma (V fragments from LDS) ----
        v16bf bv[4];
        #pragma unroll
        for (int d = 0; d < 4; ++d) {
            int col = d * 16 + lo;
            int kr  = hi ? 8 : 0;
            #pragma unroll
            for (int e = 0; e < 8; ++e) {
                bv[d][e] = Vt[buf][kr + e][col];
                bv[d][8 + e] = Vt[buf][kr + 16 + e][col];
            }
        }
        #pragma unroll
        for (int d = 0; d < 4; ++d)
            o[d] = __builtin_amdgcn_wmma_f32_16x16x32_bf16(
                false, pa, false, bv[d], (short)0, o[d], false, false);
    }

    // ---- normalize and write attn output (bf16, same layout as Q) ----
    __bf16* Ob = O + ((size_t)(b * TSEQ + q0)) * F_DIM + headOff;
    #pragma unroll
    for (int d = 0; d < 4; ++d) {
        #pragma unroll
        for (int r = 0; r < 8; ++r) {
            int row = r + (hi ? 8 : 0);
            float val = o[d][r] / l[r];
            Ob[(size_t)row * F_DIM + d * 16 + lo] = (__bf16)val;
        }
    }
}

// ---------------------------------------------------------------------------
// Launch
// ---------------------------------------------------------------------------
extern "C" void kernel_launch(void* const* d_in, const int* in_sizes, int n_in,
                              void* d_out, int out_size, void* d_ws, size_t ws_size,
                              hipStream_t stream) {
    const float* x     = (const float*)d_in[0];
    const float* ln1_g = (const float*)d_in[1];
    const float* ln1_b = (const float*)d_in[2];
    const float* Wq    = (const float*)d_in[3];
    const float* bq    = (const float*)d_in[4];
    const float* Wk    = (const float*)d_in[5];
    const float* bk    = (const float*)d_in[6];
    const float* Wv    = (const float*)d_in[7];
    const float* bv    = (const float*)d_in[8];
    const float* Wo    = (const float*)d_in[9];
    const float* bo    = (const float*)d_in[10];
    const float* ln2_g = (const float*)d_in[11];
    const float* ln2_b = (const float*)d_in[12];
    const float* W1    = (const float*)d_in[13];
    const float* b1    = (const float*)d_in[14];
    const float* W2    = (const float*)d_in[15];
    const float* b2    = (const float*)d_in[16];

    // Workspace carve-up (256B aligned)
    char* p = (char*)d_ws;
    auto alloc = [&](size_t bytes) -> char* {
        char* r = p;
        p += (bytes + 255) & ~(size_t)255;
        return r;
    };
    __bf16* hb   = (__bf16*)alloc((size_t)NTOK * F_DIM * 2);
    __bf16* Wqt  = (__bf16*)alloc((size_t)F_DIM * F_DIM * 2);
    __bf16* Wkt  = (__bf16*)alloc((size_t)F_DIM * F_DIM * 2);
    __bf16* Wvt  = (__bf16*)alloc((size_t)F_DIM * F_DIM * 2);
    __bf16* Wot  = (__bf16*)alloc((size_t)F_DIM * F_DIM * 2);
    __bf16* W1t  = (__bf16*)alloc((size_t)F_DIM * MLPH * 2);
    __bf16* W2t  = (__bf16*)alloc((size_t)MLPH * F_DIM * 2);
    __bf16* Qb   = (__bf16*)alloc((size_t)NTOK * F_DIM * 2);
    __bf16* Kb   = (__bf16*)alloc((size_t)NTOK * F_DIM * 2);
    __bf16* Vb   = (__bf16*)alloc((size_t)NTOK * F_DIM * 2);
    __bf16* Ab   = (__bf16*)alloc((size_t)NTOK * F_DIM * 2);   // attention out
    float*  outf = (float*) alloc((size_t)NTOK * F_DIM * 4);   // residual stream
    __bf16* h2b  = (__bf16*)alloc((size_t)NTOK * F_DIM * 2);
    __bf16* m1b  = (__bf16*)alloc((size_t)NTOK * MLPH * 2);

    // 1) weights -> bf16 transposed [N,K]
    {
        int nFF = F_DIM * F_DIM;
        int gFF = (nFF + 255) / 256;
        k_cast_transpose<<<gFF, 256, 0, stream>>>(Wq, Wqt, F_DIM, F_DIM);
        k_cast_transpose<<<gFF, 256, 0, stream>>>(Wk, Wkt, F_DIM, F_DIM);
        k_cast_transpose<<<gFF, 256, 0, stream>>>(Wv, Wvt, F_DIM, F_DIM);
        k_cast_transpose<<<gFF, 256, 0, stream>>>(Wo, Wot, F_DIM, F_DIM);
        int nFM = F_DIM * MLPH;
        int gFM = (nFM + 255) / 256;
        k_cast_transpose<<<gFM, 256, 0, stream>>>(W1, W1t, F_DIM, MLPH);
        k_cast_transpose<<<gFM, 256, 0, stream>>>(W2, W2t, MLPH, F_DIM);
    }

    // 2) LN1
    k_layernorm<<<NTOK, 256, 0, stream>>>(x, ln1_g, ln1_b, hb, F_DIM);

    // 3) Q,K,V projections (bf16 out)
    {
        dim3 grid(F_DIM / 256, NTOK / 128);
        k_gemm<false, false, true><<<grid, 256, 0, stream>>>(hb, Wqt, bq, nullptr, Qb, NTOK, F_DIM, F_DIM);
        k_gemm<false, false, true><<<grid, 256, 0, stream>>>(hb, Wkt, bk, nullptr, Kb, NTOK, F_DIM, F_DIM);
        k_gemm<false, false, true><<<grid, 256, 0, stream>>>(hb, Wvt, bv, nullptr, Vb, NTOK, F_DIM, F_DIM);
    }

    // 4) attention
    {
        dim3 grid(TSEQ / 64, BATCH * HEADS);
        k_attn<<<grid, 128, 0, stream>>>(Qb, Kb, Vb, Ab);
    }

    // 5) output projection + residual (fp32 out -> residual stream)
    {
        dim3 grid(F_DIM / 256, NTOK / 128);
        k_gemm<false, true, false><<<grid, 256, 0, stream>>>(Ab, Wot, bo, x, outf, NTOK, F_DIM, F_DIM);
    }

    // 6) LN2
    k_layernorm<<<NTOK, 256, 0, stream>>>(outf, ln2_g, ln2_b, h2b, F_DIM);

    // 7) MLP up + GELU (bf16 out)
    {
        dim3 grid(MLPH / 256, NTOK / 128);
        k_gemm<true, false, true><<<grid, 256, 0, stream>>>(h2b, W1t, b1, nullptr, m1b, NTOK, MLPH, F_DIM);
    }

    // 8) MLP down + GELU + residual (fp32 out -> d_out)
    {
        dim3 grid(F_DIM / 256, NTOK / 128);
        k_gemm<true, true, false><<<grid, 256, 0, stream>>>(m1b, W2t, b2, outf, (float*)d_out, NTOK, F_DIM, MLPH);
    }
}
